// Conv_39333310497378
// MI455X (gfx1250) — compile-verified, matrix operand order checked by reading, not run on hardware
//
#include <hip/hip_runtime.h>
#include <hip/hip_bf16.h>

typedef __attribute__((ext_vector_type(16))) _Float16 v16h;
typedef __attribute__((ext_vector_type(8)))  _Float16 v8h;
typedef __attribute__((ext_vector_type(4)))  _Float16 v4h;
typedef __attribute__((ext_vector_type(8)))  float    v8f;
typedef __attribute__((ext_vector_type(4)))  float    v4f;

#define NSEC   9
#define NCH    50
#define KREAL  240     // 6*40 window
#define NCHUNK 8       // Kpad 256 / 32
#define XROW   1640    // 41*40 floats per image
#define LROW   1664    // f16 halves per image row in LDS (24 zero-pad, keeps 16B align)
#define TILE_M 16
#define THRESH 6.2f

// ---- Pack W[9,50,1,6,40] f32 into WMMA-B-fragment-native f16 layout ----
// wp[blk][lane][16] halves, blk = (s*4 + ntile)*8 + chunk.
// B frag (f16, 32x16 KxN): lane<16 -> N=lane, K = {k0..k0+7, k0+16..k0+23}
//                          lane>=16 -> N=lane-16, K groups shifted by +8.
__global__ __launch_bounds__(32)
void pack_w_kernel(const float* __restrict__ W, _Float16* __restrict__ wp) {
  const int blk   = blockIdx.x;
  const int chunk = blk & 7;
  const int nt    = (blk >> 3) & 3;
  const int s     = blk >> 5;
  const int lane  = threadIdx.x;
  const int o     = nt * 16 + (lane & 15);
  const int kbase = chunk * 32 + ((lane & 16) ? 8 : 0);
  const float* wsrc = W + (size_t)(s * NCH + o) * KREAL;
  v16h frag;
#pragma unroll
  for (int i = 0; i < 16; ++i) {
    const int k = kbase + (i & 7) + ((i >> 3) * 16);
    float v = 0.0f;
    if (o < NCH && k < KREAL) v = wsrc[k];
    frag[i] = (_Float16)v;
  }
  *(v16h*)(wp + (size_t)blk * 512 + lane * 16) = frag;
}

__device__ __forceinline__ v16h load_a_frag(const _Float16* ap) {
  v8h g0 = *(const v8h*)ap;          // K group 0 (8 halves, 16B aligned: ds_load_b128)
  v8h g1 = *(const v8h*)(ap + 16);   // K group 1
  return __builtin_shufflevector(g0, g1, 0,1,2,3,4,5,6,7,8,9,10,11,12,13,14,15);
}

__global__ __launch_bounds__(128)
void snn_conv_kernel(const float* __restrict__ x, const _Float16* __restrict__ wp,
                     float* __restrict__ out, int batch) {
  __shared__ _Float16 lx[TILE_M * LROW];   // 52 KB

  const int tid   = threadIdx.x;
  const int tileb = blockIdx.x * TILE_M;

  // ---- stage 16 images into LDS as f16 (contiguous f32 stream -> b128 loads) ----
  const float* src = x + (size_t)tileb * XROW;
  for (int i = tid; i < TILE_M * (XROW / 4); i += 128) {
    const int r  = i / (XROW / 4);
    const int c4 = i - r * (XROW / 4);
    v4f v = *(const v4f*)(src + (size_t)r * XROW + c4 * 4);
    v4h h;
    h.x = (_Float16)v.x; h.y = (_Float16)v.y;
    h.z = (_Float16)v.z; h.w = (_Float16)v.w;
    *(v4h*)(lx + r * LROW + c4 * 4) = h;   // 8B-aligned ds_store_b64
  }
  // zero the per-row pad so padded-K A reads are 0 (never NaN garbage; 0*0=0 in WMMA)
  for (int i = tid; i < TILE_M * (LROW - XROW); i += 128) {
    const int r = i / (LROW - XROW);
    const int c = i - r * (LROW - XROW);
    lx[r * LROW + XROW + c] = (_Float16)0.0f;
  }
  __syncthreads();

  const int lane  = tid & 31;
  const int ntile = tid >> 5;        // each wave owns one 16-channel tile
  const int hi    = (lane >> 4) & 1; // K-group half select per A/B striping
  const int ocol  = lane & 15;

  // A: M = lane&15 (batch row within tile) for both lane halves; hi shifts K by +8
  const _Float16* arow = lx + (lane & 15) * LROW + hi * 8;

  float* pots = out;
  float* spks = out + (size_t)batch * NCH * NSEC;

  for (int s = 0; s < NSEC; ++s) {
    v8f acc0 = {}, acc1 = {}, acc2 = {}, acc3 = {};   // one accumulator per conv offset h
    const _Float16* bsec = wp + (size_t)(s * 4 + ntile) * (NCHUNK * 512) + lane * 16;

    for (int chunk = 0; chunk < NCHUNK; ++chunk) {
      v16h bf = *(const v16h*)(bsec + (size_t)chunk * 512);  // 32B contiguous, reused 4x
      const int kb = s * 4 * 40 + chunk * 32;
      v16h a0 = load_a_frag(arow + kb + 0 * 40);
      v16h a1 = load_a_frag(arow + kb + 1 * 40);
      v16h a2 = load_a_frag(arow + kb + 2 * 40);
      v16h a3 = load_a_frag(arow + kb + 3 * 40);
      acc0 = __builtin_amdgcn_wmma_f32_16x16x32_f16(false, a0, false, bf, (short)0, acc0, false, false);
      acc1 = __builtin_amdgcn_wmma_f32_16x16x32_f16(false, a1, false, bf, (short)0, acc1, false, false);
      acc2 = __builtin_amdgcn_wmma_f32_16x16x32_f16(false, a2, false, bf, (short)0, acc2, false, false);
      acc3 = __builtin_amdgcn_wmma_f32_16x16x32_f16(false, a3, false, bf, (short)0, acc3, false, false);
    }

    // max-pool over the 4 h offsets, threshold, scatter-store
    const int o = ntile * 16 + ocol;
    if (o < NCH) {
      const int browbase = tileb + hi * 8;   // D: VGPR r -> M = r + 8*hi
#pragma unroll
      for (int r = 0; r < 8; ++r) {
        const float p = fmaxf(fmaxf(acc0[r], acc1[r]), fmaxf(acc2[r], acc3[r]));
        const size_t idx = (size_t)(browbase + r) * (NCH * NSEC) + (size_t)o * NSEC + s;
        pots[idx] = p;
        spks[idx] = (p > THRESH) ? 1.0f : 0.0f;
      }
    }
  }
}

extern "C" void kernel_launch(void* const* d_in, const int* in_sizes, int n_in,
                              void* d_out, int out_size, void* d_ws, size_t ws_size,
                              hipStream_t stream) {
  const float* x = (const float*)d_in[0];   // [B,1,41,40] f32
  const float* W = (const float*)d_in[1];   // [9,50,1,6,40] f32
  float* out = (float*)d_out;               // pots [B,50,9] ++ spks [B,50,9]
  _Float16* wp = (_Float16*)d_ws;           // 294912 B of packed W fragments

  const int batch = in_sizes[0] / XROW;     // 16384

  pack_w_kernel<<<NSEC * 4 * NCHUNK, 32, 0, stream>>>(W, wp);
  snn_conv_kernel<<<batch / TILE_M, 128, 0, stream>>>(x, wp, out, batch);
}
